// GCPNet_42786464203393
// MI455X (gfx1250) — matrix operand order repaired and checked by reading, not compiled, not run on hardware
//
#include <hip/hip_runtime.h>
#include <cstdint>
#include <cstddef>

// ---------------------------------------------------------------------------
// GCPNet on MI455X (gfx1250, wave32).
//
// - All GEMMs: bf16 x bf16 -> f32 via v_wmma_f32_16x16x32_bf16.
//   M_vv (32MB as bf16) is L2-resident (192MB L2); ~85 GFLOP total is far
//   below the WMMA roof -> GEMM is latency/L2-bound. K loop: 2x unrolled
//   ping-pong, last chunk-pair peeled into an epilogue so the loop body is
//   branch-free straight-line code (loads unconditional) -> the compiler can
//   emit partial s_wait_loadcnt instead of full drains; each set's
//   load->use distance is a full iteration of compute.
// - The serial LSTM scan (32*4096 + 32*128 dependent steps) dominates wall
//   time: Whh rows pinned in VGPRs, h read as float4 (ds_load_b128, 16/step),
//   xg[t+1] prefetched before step t's barriers, 2 barriers/step, state
//   persisted across outer iterations.
// - 3-layer MLPs fused per 16-row tile; inter-layer tiles re-staged through
//   wave-private LDS (per-wave DS ordering => barrier-free).
// ---------------------------------------------------------------------------

typedef __bf16 bf16_t;
typedef __attribute__((ext_vector_type(16))) __bf16 v16bf;
typedef __attribute__((ext_vector_type(8)))  __bf16 v8bf;
typedef __attribute__((ext_vector_type(8)))  float  v8f;

#define D_   64
#define N_   4096
#define NC_  128
#define TMAX_ 32

#define SHUF16(lo, hi) __builtin_shufflevector(lo, hi, 0,1,2,3,4,5,6,7,8,9,10,11,12,13,14,15)

// ---------------------------------------------------------------------------
// Generic bf16 WMMA GEMM:  D[M,N] = A[M,K] @ Bt[N,K]^T  (+bias, optional relu)
// A row-major bf16 (lda), Bt row-major bf16 (ldb): contiguous-K 16B loads.
// Outputs (any subset): f32 row-major, bf16 row-major, bf16 transposed [N,M].
// Each wave: 16(M) x 64(N) strip, 4 accumulators.
// K loop: ping-pong with peeled epilogue (branch-free body, unconditional
// loads). Grid: (M/16)*(N/64) waves, 8 waves per block. K % 64 == 0.
// ---------------------------------------------------------------------------
__global__ __launch_bounds__(256) void gemm_bf16_wmma(
    const bf16_t* __restrict__ A, int lda,
    const bf16_t* __restrict__ Bt, int ldb,
    const float* __restrict__ bias,
    float*  __restrict__ outF, int ldof,
    bf16_t* __restrict__ outB, int ldob,
    bf16_t* __restrict__ outT, int ldot,
    int N, int K, int relu)
{
  const int wave = (blockIdx.x * 256 + threadIdx.x) >> 5;
  const int lane = threadIdx.x & 31;
  const int nStrips = N >> 6;
  const int m0 = (wave / nStrips) << 4;
  const int n0 = (wave % nStrips) << 6;
  const int lh = lane & 15;
  const int hi = lane >> 4;

  // A (16x32 bf16): lanes 0-15 K {0..7,16..23}; lanes 16-31 K {8..15,24..31}
  const int koffA = hi << 3;          // second chunk at +16
  // B (32x16 bf16): lanes 0-15 K {0..7,16..23}; lanes 16-31 K {8..15,24..31}
  const int koffB = hi << 3;          // second chunk at +16

  const bf16_t* aRow  = A  + (size_t)(m0 + lh) * lda + koffA;
  const bf16_t* bBase = Bt + (size_t)(n0 + lh) * ldb + koffB;

  v8f acc[4];
#pragma unroll
  for (int nb = 0; nb < 4; ++nb)
#pragma unroll
    for (int r = 0; r < 8; ++r) acc[nb][r] = 0.0f;

#define LOAD_TILES(aL, aH, bL, bH, kk)                                    \
  {                                                                       \
    aL = *(const v8bf*)(aRow + (kk));                                     \
    aH = *(const v8bf*)(aRow + (kk) + 16);                                \
    _Pragma("unroll")                                                     \
    for (int nb_ = 0; nb_ < 4; ++nb_) {                                   \
      const bf16_t* bRow = bBase + (size_t)(nb_ * 16) * ldb + (kk);       \
      bL[nb_] = *(const v8bf*)(bRow);                                     \
      bH[nb_] = *(const v8bf*)(bRow + 16);                                \
    }                                                                     \
  }

#define GEMM_STEP(aL, aH, bL, bH)                                         \
  {                                                                       \
    v16bf a_ = SHUF16(aL, aH);                                            \
    _Pragma("unroll")                                                     \
    for (int nb_ = 0; nb_ < 4; ++nb_) {                                   \
      v16bf b_ = SHUF16(bL[nb_], bH[nb_]);                                \
      acc[nb_] = __builtin_amdgcn_wmma_f32_16x16x32_bf16(                 \
          false, a_, false, b_, (short)0, acc[nb_], false, false);        \
    }                                                                     \
  }

  // -------- ping-pong K loop with peeled epilogue (K % 64 == 0) --------
  // Loop body is branch-free: all loads unconditional, each set reloaded
  // right after consumption (load->use distance = one full iteration).
  v8bf a0L, a0H, b0L[4], b0H[4];
  v8bf a1L, a1H, b1L[4], b1H[4];
  LOAD_TILES(a0L, a0H, b0L, b0H, 0);
  LOAD_TILES(a1L, a1H, b1L, b1H, 32);
  for (int kk = 0; kk + 64 < K; kk += 64) {
    GEMM_STEP(a0L, a0H, b0L, b0H);
    LOAD_TILES(a0L, a0H, b0L, b0H, kk + 64);
    GEMM_STEP(a1L, a1H, b1L, b1H);
    LOAD_TILES(a1L, a1H, b1L, b1H, kk + 96);
  }
  GEMM_STEP(a0L, a0H, b0L, b0H);       // epilogue: last chunk pair
  GEMM_STEP(a1L, a1H, b1L, b1H);
#undef LOAD_TILES
#undef GEMM_STEP

  // D (16x16 f32): vgpr r -> M = m0 + r + 8*hi ; N = n0 + nb*16 + (lane&15)
  const int mrow = m0 + (hi << 3);
#pragma unroll
  for (int nb = 0; nb < 4; ++nb) {
    const int j = n0 + nb * 16 + lh;
    const float bs = bias ? bias[j] : 0.0f;
    float vals[8];
#pragma unroll
    for (int r = 0; r < 8; ++r) {
      float v = acc[nb][r] + bs;
      if (relu) v = v > 0.0f ? v : 0.0f;
      vals[r] = v;
      const int m = mrow + r;
      if (outF) outF[(size_t)m * ldof + j] = v;
      if (outB) outB[(size_t)m * ldob + j] = (bf16_t)v;
    }
    if (outT) {                       // 8 contiguous bf16 along M -> one b128
      v8bf pk;
#pragma unroll
      for (int r = 0; r < 8; ++r) pk[r] = (bf16_t)vals[r];
      *(v8bf*)(outT + (size_t)j * ldot + mrow) = pk;
    }
  }
}

// ---------------------------------------------------------------------------
// Fused 3-layer MLP (d->d->d, relu after layers 0,1), row-wise => fuse per
// 16-row tile. Inter-layer activations re-staged via wave-private LDS
// (per-wave DS ordering => no barriers). Final layer emits transposed bf16.
// Grid: M/16 waves, 8 waves/block.
// ---------------------------------------------------------------------------
__global__ __launch_bounds__(256) void mlp3_bf16_wmma(
    const bf16_t* __restrict__ X,      // [M,64] bf16 row-major
    const bf16_t* __restrict__ W,      // [3,64,64] bf16 (Bt per layer)
    const float* __restrict__ bias,    // [3,64]
    bf16_t* __restrict__ outT, int ldot)  // [64, M]
{
  __shared__ bf16_t tile[8][16 * 64];
  const int wid  = threadIdx.x >> 5;
  const int wave = blockIdx.x * 8 + wid;
  const int lane = threadIdx.x & 31;
  const int lh = lane & 15;
  const int hi = lane >> 4;
  const int koffA = hi << 3;
  const int koffB = hi << 3;
  const int m0 = wave << 4;
  const int mrowL = hi << 3;           // local row base inside tile
  bf16_t* my = tile[wid];

#pragma unroll
  for (int l = 0; l < 3; ++l) {
    const bf16_t* Bt = W + l * 4096;
    const float*  bs = bias + l * 64;
    v8f acc[4];
#pragma unroll
    for (int nb = 0; nb < 4; ++nb)
#pragma unroll
      for (int r = 0; r < 8; ++r) acc[nb][r] = 0.0f;

#pragma unroll
    for (int kk = 0; kk < 64; kk += 32) {
      v16bf a;
      if (l == 0) {
        const bf16_t* aRow = X + (size_t)(m0 + lh) * 64 + koffA + kk;
        a = SHUF16(*(const v8bf*)(aRow), *(const v8bf*)(aRow + 16));
      } else {
        const bf16_t* aRow = my + lh * 64 + koffA + kk;  // ds_load_b128
        a = SHUF16(*(const v8bf*)(aRow), *(const v8bf*)(aRow + 16));
      }
#pragma unroll
      for (int nb = 0; nb < 4; ++nb) {
        const bf16_t* bRow = Bt + (size_t)(nb * 16 + lh) * 64 + koffB + kk;
        v16bf b = SHUF16(*(const v8bf*)(bRow), *(const v8bf*)(bRow + 16));
        acc[nb] = __builtin_amdgcn_wmma_f32_16x16x32_bf16(
            false, a, false, b, (short)0, acc[nb], false, false);
      }
    }

    if (l < 2) {                       // bias+relu, stage into LDS (D coords)
#pragma unroll
      for (int nb = 0; nb < 4; ++nb) {
        const int j = nb * 16 + lh;
        const float b = bs[j];
#pragma unroll
        for (int r = 0; r < 8; ++r) {
          float v = acc[nb][r] + b;
          v = v > 0.0f ? v : 0.0f;
          my[(mrowL + r) * 64 + j] = (bf16_t)v;
        }
      }
    } else {                           // final: bias, transposed global store
      const int mrow = m0 + mrowL;
#pragma unroll
      for (int nb = 0; nb < 4; ++nb) {
        const int j = nb * 16 + lh;
        const float b = bs[j];
        v8bf pk;
#pragma unroll
        for (int r = 0; r < 8; ++r) pk[r] = (bf16_t)(acc[nb][r] + b);
        *(v8bf*)(outT + (size_t)j * ldot + mrow) = pk;
      }
    }
  }
}

// ---------------------------------------------------------------------------
// Dual sequential LSTM scan. Block 0: vertex LSTM (L=4096). Block 1: color
// LSTM (L=128). xg[t] = x@Wih^T + bih + bhh precomputed by GEMM. Per step:
//   g_j = xg[t,j] + sum_k Whh[j,k]*h[k]   (Whh row in VGPRs, h via float4 LDS)
//   c = sig(f)*c + sig(i)*tanh(g) ; h = sig(o)*tanh(c)
// xg[t+1] is prefetched before step t's barriers to hide VMEM latency.
// ---------------------------------------------------------------------------
__global__ __launch_bounds__(256) void lstm_dual(
    const float* __restrict__ xgV, const float* __restrict__ WhhV,
    float* __restrict__ hvS, float* __restrict__ cvS,
    float* __restrict__ VoutF, bf16_t* __restrict__ VoutB, bf16_t* __restrict__ VoutT, int LV,
    const float* __restrict__ xgC, const float* __restrict__ WhhC,
    float* __restrict__ hcS, float* __restrict__ ccS,
    float* __restrict__ CoutF, bf16_t* __restrict__ CoutB, int LC)
{
  const float *xg, *Whh; float *hS, *cS, *oF; bf16_t *oB, *oT; int L, ldT;
  if (blockIdx.x == 0) { xg=xgV; Whh=WhhV; hS=hvS; cS=cvS; oF=VoutF; oB=VoutB; oT=VoutT; L=LV; ldT=LV; }
  else                 { xg=xgC; Whh=WhhC; hS=hcS; cS=ccS; oF=CoutF; oB=CoutB; oT=nullptr; L=LC; ldT=LC; }

  __shared__ __align__(16) float sh[64];
  __shared__ float sg[256];
  const int j = threadIdx.x;

  float w[64];
#pragma unroll
  for (int k = 0; k < 64; ++k) w[k] = Whh[(size_t)j * 64 + k];

  float c = 0.0f;
  if (j < 64) { sh[j] = hS[j]; c = cS[j]; }
  __syncthreads();

  const int gi = j >> 6;              // 0:i 1:f 2:g 3:o
  const float4* sh4 = (const float4*)sh;
  float gx = xg[j];                   // prefetched input contribution (t=0)
  for (int t = 0; t < L; ++t) {
    float gnext = 0.0f;
    if (t + 1 < L) gnext = xg[(size_t)(t + 1) * 256 + j];  // hide VMEM latency
    float g = gx;
#pragma unroll
    for (int k = 0; k < 16; ++k) {    // 16x ds_load_b128 broadcast
      float4 h4 = sh4[k];
      g += w[4*k+0]*h4.x + w[4*k+1]*h4.y + w[4*k+2]*h4.z + w[4*k+3]*h4.w;
    }
    float a = (gi == 2) ? tanhf(g) : (1.0f / (1.0f + __expf(-g)));
    sg[j] = a;
    __syncthreads();
    if (j < 64) {
      c = sg[64 + j] * c + sg[j] * sg[128 + j];
      float h = sg[192 + j] * tanhf(c);
      sh[j] = h;
      oF[(size_t)t * 64 + j] = h;
      oB[(size_t)t * 64 + j] = (bf16_t)h;
      if (oT) oT[(size_t)j * ldT + t] = (bf16_t)h;
    }
    __syncthreads();
    gx = gnext;
  }
  if (j < 64) { hS[j] = sh[j]; cS[j] = c; }
}

// ---------------------------------------------------------------------------
// Prep / epilogue helpers
// ---------------------------------------------------------------------------
__global__ __launch_bounds__(256) void cvt_bf16(const float* __restrict__ s,
                                                bf16_t* __restrict__ d, long n) {
  long i = (long)blockIdx.x * 256 + threadIdx.x;
  const long stride = (long)gridDim.x * 256;
  for (; i < n; i += stride) d[i] = (bf16_t)s[i];
}

__global__ __launch_bounds__(256) void transpose_cvt(const float* __restrict__ s,
                                                     bf16_t* __restrict__ d, int R, int C) {
  long i = (long)blockIdx.x * 256 + threadIdx.x;
  const long n = (long)R * C;
  const long stride = (long)gridDim.x * 256;
  for (; i < n; i += stride) {
    int r = (int)(i / C), c = (int)(i % C);
    d[(long)c * R + r] = (bf16_t)s[i];
  }
}

__global__ __launch_bounds__(256) void prep_misc(
    const float* bihv, const float* bhhv, const float* bihc, const float* bhhc,
    const float* Vh0, const float* Ch0,
    float* biasV, float* biasC, float* hv, float* cv, float* hc, float* cc) {
  int j = threadIdx.x;
  biasV[j] = bihv[j] + bhhv[j];
  biasC[j] = bihc[j] + bhhc[j];
  if (j < 64) { hv[j] = Vh0[j]; cv[j] = 0.0f; hc[j] = Ch0[j]; cc[j] = 0.0f; }
}

// vote[n] = h2[n]·Wo + bo ; votes[g] = sigmoid(mean over 128 nodes)
__global__ __launch_bounds__(128) void vote_final(const bf16_t* __restrict__ h2,
                                                  const float* __restrict__ Wo,
                                                  const float* __restrict__ bo,
                                                  float* __restrict__ out) {
  __shared__ float red[128];
  const int g = blockIdx.x, t = threadIdx.x;
  const int n = g * 128 + t;
  float s = bo[0];
#pragma unroll
  for (int k = 0; k < 64; ++k) s += (float)h2[(size_t)n * 64 + k] * Wo[k];
  red[t] = s;
  __syncthreads();
  for (int st = 64; st > 0; st >>= 1) {
    if (t < st) red[t] += red[t + st];
    __syncthreads();
  }
  if (t == 0) out[g] = 1.0f / (1.0f + __expf(-red[0] / 128.0f));
}

// ---------------------------------------------------------------------------
extern "C" void kernel_launch(void* const* d_in, const int* in_sizes, int n_in,
                              void* d_out, int out_size, void* d_ws, size_t ws_size,
                              hipStream_t stream) {
  const float* M_vv      = (const float*)d_in[0];
  const float* M_vc      = (const float*)d_in[1];
  const float* V_in      = (const float*)d_in[2];
  const float* C_in      = (const float*)d_in[3];
  const float* mlpC_Ws   = (const float*)d_in[4];
  const float* mlpC_bs   = (const float*)d_in[5];
  const float* mlpV_Ws   = (const float*)d_in[6];
  const float* mlpV_bs   = (const float*)d_in[7];
  const float* vote_Ws   = (const float*)d_in[8];
  const float* vote_bs   = (const float*)d_in[9];
  const float* vote_Wo   = (const float*)d_in[10];
  const float* vote_bo   = (const float*)d_in[11];
  const float* lv_Wih    = (const float*)d_in[12];
  const float* lv_Whh    = (const float*)d_in[13];
  const float* lv_bih    = (const float*)d_in[14];
  const float* lv_bhh    = (const float*)d_in[15];
  const float* lc_Wih    = (const float*)d_in[16];
  const float* lc_Whh    = (const float*)d_in[17];
  const float* lc_bih    = (const float*)d_in[18];
  const float* lc_bhh    = (const float*)d_in[19];
  const float* V_h0      = (const float*)d_in[20];
  const float* C_h0      = (const float*)d_in[21];
  (void)in_sizes; (void)n_in; (void)out_size;

  float* out   = (float*)d_out;
  float* votes = out;                        // [32]
  float* Vf    = out + 32;                   // [4096*64]
  float* Cf    = out + 32 + N_ * D_;         // [128*64]

  char* ws = (char*)d_ws; size_t off = 0;
  auto alloc = [&](size_t bytes) -> void* {
    void* p = ws + off; off = (off + bytes + 255) & ~(size_t)255; return p;
  };
  bf16_t* Mvv_bf   = (bf16_t*)alloc((size_t)N_ * N_ * 2);     // 32MB, L2-resident
  bf16_t* Mvc_bf   = (bf16_t*)alloc((size_t)N_ * NC_ * 2);
  bf16_t* MvcT_bf  = (bf16_t*)alloc((size_t)N_ * NC_ * 2);    // [128,4096]
  bf16_t* Wihv_bf  = (bf16_t*)alloc(256 * 128 * 2);
  bf16_t* Wihc_bf  = (bf16_t*)alloc(256 * 64 * 2);
  bf16_t* mlpCW_bf = (bf16_t*)alloc(3 * 64 * 64 * 2);
  bf16_t* mlpVW_bf = (bf16_t*)alloc(3 * 64 * 64 * 2);
  bf16_t* voteW_bf = (bf16_t*)alloc(2 * 64 * 64 * 2);
  bf16_t* V_bf     = (bf16_t*)alloc((size_t)N_ * D_ * 2);     // [4096,64]
  bf16_t* Vt_bf    = (bf16_t*)alloc((size_t)N_ * D_ * 2);     // [64,4096]
  bf16_t* C_bf     = (bf16_t*)alloc((size_t)NC_ * D_ * 2);
  bf16_t* mlpCt_bf = (bf16_t*)alloc((size_t)NC_ * D_ * 2);    // [64,128]
  bf16_t* mlpVt_bf = (bf16_t*)alloc((size_t)N_ * D_ * 2);     // [64,4096]
  bf16_t* xv_bf    = (bf16_t*)alloc((size_t)N_ * 128 * 2);    // [4096,128]
  bf16_t* xc_bf    = (bf16_t*)alloc((size_t)NC_ * D_ * 2);    // [128,64]
  float*  xg_v     = (float*)alloc((size_t)N_ * 256 * 4);     // [4096,256]
  float*  xg_c     = (float*)alloc((size_t)NC_ * 256 * 4);
  float*  biasV    = (float*)alloc(256 * 4);
  float*  biasC    = (float*)alloc(256 * 4);
  float*  hv = (float*)alloc(64 * 4); float* cv = (float*)alloc(64 * 4);
  float*  hc = (float*)alloc(64 * 4); float* cc = (float*)alloc(64 * 4);
  bf16_t* h1_bf    = (bf16_t*)alloc((size_t)N_ * D_ * 2);
  bf16_t* h2_bf    = (bf16_t*)alloc((size_t)N_ * D_ * 2);
  (void)ws_size;

  // ---- one-time prep (re-done every call: stateless & deterministic) ----
  cvt_bf16<<<4096, 256, 0, stream>>>(M_vv, Mvv_bf, (long)N_ * N_);
  cvt_bf16<<<512,  256, 0, stream>>>(M_vc, Mvc_bf, (long)N_ * NC_);
  transpose_cvt<<<512, 256, 0, stream>>>(M_vc, MvcT_bf, N_, NC_);
  cvt_bf16<<<128, 256, 0, stream>>>(lv_Wih, Wihv_bf, 256 * 128);
  cvt_bf16<<<64,  256, 0, stream>>>(lc_Wih, Wihc_bf, 256 * 64);
  cvt_bf16<<<48,  256, 0, stream>>>(mlpC_Ws, mlpCW_bf, 3 * 64 * 64);
  cvt_bf16<<<48,  256, 0, stream>>>(mlpV_Ws, mlpVW_bf, 3 * 64 * 64);
  cvt_bf16<<<32,  256, 0, stream>>>(vote_Ws, voteW_bf, 2 * 64 * 64);
  cvt_bf16<<<512, 256, 0, stream>>>(V_in, V_bf, (long)N_ * D_);
  transpose_cvt<<<512, 256, 0, stream>>>(V_in, Vt_bf, N_, D_);
  cvt_bf16<<<32, 256, 0, stream>>>(C_in, C_bf, (long)NC_ * D_);
  prep_misc<<<1, 256, 0, stream>>>(lv_bih, lv_bhh, lc_bih, lc_bhh, V_h0, C_h0,
                                   biasV, biasC, hv, cv, hc, cc);

  auto gemm = [&](const bf16_t* A, int lda, const bf16_t* Bt, int ldb,
                  const float* bias, float* oF, int ldof,
                  bf16_t* oB, int ldob, bf16_t* oT, int ldot,
                  int M, int N, int K, int relu) {
    int waves = (M / 16) * (N / 64);           // all call sites: waves % 8 == 0
    gemm_bf16_wmma<<<waves / 8, 256, 0, stream>>>(A, lda, Bt, ldb, bias,
                                                  oF, ldof, oB, ldob, oT, ldot,
                                                  N, K, relu);
  };

  for (int t = 0; t < TMAX_; ++t) {
    // mlpC(C) -> transposed bf16 [64,128] ; mlpV(V) -> transposed bf16 [64,4096]
    mlp3_bf16_wmma<<<1, 256, 0, stream>>>(C_bf, mlpCW_bf, mlpC_bs, mlpCt_bf, 128);
    mlp3_bf16_wmma<<<32, 256, 0, stream>>>(V_bf, mlpVW_bf, mlpV_bs, mlpVt_bf, N_);
    // xv = [ M_vv@V | M_vc@mlpC(C) ]  -> bf16 [4096,128]
    gemm(Mvv_bf, N_,  Vt_bf,    N_,  nullptr, nullptr,0, xv_bf,     128, nullptr,0, N_, 64, N_,  0);
    gemm(Mvc_bf, NC_, mlpCt_bf, NC_, nullptr, nullptr,0, xv_bf + 64,128, nullptr,0, N_, 64, NC_, 0);
    // xc = M_vc^T @ mlpV(V)  -> bf16 [128,64]
    gemm(MvcT_bf, N_, mlpVt_bf, N_, nullptr, nullptr,0, xc_bf,64, nullptr,0, NC_, 64, N_, 0);
    // xg = x @ Wih^T + (bih+bhh)  -> f32
    gemm(xv_bf, 128, Wihv_bf, 128, biasV, xg_v,256, nullptr,0, nullptr,0, N_,  256, 128, 0);
    gemm(xc_bf, 64,  Wihc_bf, 64,  biasC, xg_c,256, nullptr,0, nullptr,0, NC_, 256, 64,  0);
    // serial LSTM scans (block 0: vertices, block 1: colors)
    lstm_dual<<<2, 256, 0, stream>>>(xg_v, lv_Whh, hv, cv, Vf, V_bf, Vt_bf, N_,
                                     xg_c, lc_Whh, hc, cc, Cf, C_bf, NC_);
  }

  // vote head: two relu'd 64x64 layers (WMMA) + dot/mean/sigmoid
  gemm(V_bf,  64, voteW_bf,        64, vote_bs,      nullptr,0, h1_bf,64, nullptr,0, N_, 64, 64, 1);
  gemm(h1_bf, 64, voteW_bf + 4096, 64, vote_bs + 64, nullptr,0, h2_bf,64, nullptr,0, N_, 64, 64, 1);
  vote_final<<<32, 128, 0, stream>>>(h2_bf, vote_Wo, vote_bo, votes);
}
